// RENODE_89962384982402
// MI455X (gfx1250) — compile-verified, matrix operand order checked by reading, not run on hardware
//
#include <hip/hip_runtime.h>
#include <math.h>

#define NBATCH 8192
#define NXD 512
#define NYD 256
#define NUD 256
#define NQD 512

#define BM 32            // batch rows per workgroup
#define PP (NQD + 4)     // pre pitch (f32)
#define PX (NXD + 8)     // x tile pitch (f16)
#define PU (NUD + 8)     // u tile pitch (f16)
#define PW (NQD + 8)     // w tile pitch (f16)

typedef __attribute__((ext_vector_type(16))) _Float16 v16h;
typedef __attribute__((ext_vector_type(8)))  _Float16 v8h;
typedef __attribute__((ext_vector_type(8)))  float    v8f;

// ---- WMMA wrapper: D = A(16x32 f16) x B(32x16 f16) + C(16x16 f32) ----
__device__ __forceinline__ v8f wmma_f16(v16h a, v16h b, v8f c) {
  return __builtin_amdgcn_wmma_f32_16x16x32_f16(
      /*neg_a=*/false, a, /*neg_b=*/false, b,
      /*c_mod=*/(short)0, c, /*reuse_a=*/false, /*reuse_b=*/false);
}

// ---- Load one 16x32 f16 fragment (A role, or B role with row = output col).
// ISA layout: lanes 0-15 hold K=0..7 (v0-3) and K=16..23 (v4-7);
// lanes 16-31 hold K=8..15 and K=24..31. Two 16-byte loads per lane. ----
__device__ __forceinline__ v16h load_frag16x32(const _Float16* base, int rowStride,
                                               int row, int kb, int lane) {
  const int khalf = (lane >> 4) * 8;
  const _Float16* p = base + (size_t)row * rowStride + kb + khalf;
  v8h lo = *(const v8h*)(p);
  v8h hi = *(const v8h*)(p + 16);
  return __builtin_shufflevector(lo, hi, 0,1,2,3,4,5,6,7,8,9,10,11,12,13,14,15);
}

// ---- accumulate acc += Atile(BM-row tile) * Wrow-majorᵀ over K ----
__device__ __forceinline__ v8f gemm_acc(v8f acc,
                                        const _Float16* aBase, int aStride, int aRow,
                                        const _Float16* bBase, int bStride, int bRow,
                                        int K, int lane) {
  for (int kb = 0; kb < K; kb += 32) {
    v16h a = load_frag16x32(aBase, aStride, aRow, kb, lane);
    v16h b = load_frag16x32(bBase, bStride, bRow, kb, lane);
    acc = wmma_f16(a, b, acc);
  }
  return acc;
}

// ---- f32 -> f16 weight conversion ----
__global__ void ren_cvt_f16(const float* __restrict__ s, _Float16* __restrict__ d, int n) {
  int i = blockIdx.x * blockDim.x + threadIdx.x;
  if (i < n) d[i] = (_Float16)s[i];
}

// ---- fused REN forward: pre-GEMM, blocked tanh-triangular solve, out-GEMMs ----
__global__ __launch_bounds__(256)
void ren_fused(const float* __restrict__ u,  const float* __restrict__ x,
               const _Float16* __restrict__ C1h,  const _Float16* __restrict__ D12h,
               const _Float16* __restrict__ D11h, const _Float16* __restrict__ Ah,
               const _Float16* __restrict__ B1h,  const _Float16* __restrict__ B2h,
               const _Float16* __restrict__ C2h,  const _Float16* __restrict__ D21h,
               const _Float16* __restrict__ D22h,
               const float* __restrict__ D11f, const float* __restrict__ Lam,
               const float* __restrict__ bv,
               float* __restrict__ dx_out, float* __restrict__ y_out) {
  extern __shared__ char smem[];
  float*    pre  = (float*)(smem);                    // 32*516*4 = 66048 B
  _Float16* xs   = (_Float16*)(smem + 66048);         // 32*520*2 = 33280 B
  _Float16* us   = (_Float16*)(smem + 99328);         // 32*264*2 = 16896 B
  _Float16* wsm  = (_Float16*)(smem + 116224);        // 32*520*2 = 33280 B
  float*    d11b = (float*)(smem + 149504);           // 32*32*4  =  4096 B
  float*    lamb = (float*)(smem + 153600);           // 32*4     =   128 B

  const int tid  = threadIdx.x;
  const int lane = tid & 31;
  const int wid  = tid >> 5;
  const int ln   = lane & 15;          // fragment row/col within tile
  const int rh   = (lane >> 4) * 8;    // C/D row-half offset
  const int b0   = blockIdx.x * BM;

  // ---- stage x,u tiles into LDS as f16 ----
  for (int i = tid; i < BM * NXD; i += 256) {
    int r = i >> 9, c = i & (NXD - 1);
    xs[r * PX + c] = (_Float16)x[(size_t)(b0 + r) * NXD + c];
  }
  for (int i = tid; i < BM * NUD; i += 256) {
    int r = i >> 8, c = i & (NUD - 1);
    us[r * PU + c] = (_Float16)u[(size_t)(b0 + r) * NUD + c];
  }
  __syncthreads();

  // ---- pre = x@C1^T + u@D12^T + bv  (2 m-tiles x 32 q-tiles over 8 waves) ----
  for (int tile = wid; tile < 64; tile += 8) {
    const int mt = tile & 1, qt = tile >> 1;
    v8f acc = {};
    acc = gemm_acc(acc, xs, PX, mt * 16 + ln, C1h,  NXD, qt * 16 + ln, NXD, lane);
    acc = gemm_acc(acc, us, PU, mt * 16 + ln, D12h, NUD, qt * 16 + ln, NUD, lane);
    const float bvv = bv[qt * 16 + ln];
    const int rowb = mt * 16 + rh;
#pragma unroll
    for (int r = 0; r < 8; ++r) pre[(rowb + r) * PP + qt * 16 + ln] = acc[r] + bvv;
  }

  // ---- blocked forward substitution: 16 blocks of 32 columns ----
  for (int t = 0; t < 16; ++t) {
    const int kb = t * 32;
    // stage D11 diagonal block + 1/Lambda (all threads)
    for (int i = tid; i < 32 * 32; i += 256) {
      int kk = i >> 5, j = i & 31;
      d11b[i] = D11f[(size_t)(kb + kk) * NQD + kb + j];
    }
    if (tid < 32) lamb[tid] = 1.0f / Lam[kb + tid];
    __syncthreads();   // pre(trailing t-1) + stage visible before solve

    if (wid == 0) {    // scalar triangular recurrence; lane = batch row
      const int b = lane;
      float wl[32];
#pragma unroll
      for (int kk = 0; kk < 32; ++kk) {
        float s = pre[b * PP + kb + kk];
#pragma unroll
        for (int j = 0; j < kk; ++j) s = fmaf(wl[j], d11b[kk * 32 + j], s);
        const float v = tanhf(s * lamb[kk]);
        wl[kk] = v;
        wsm[b * PW + kb + kk] = (_Float16)v;
      }
    }
    __syncthreads();   // w block visible before trailing update

    // rank-32 trailing update: pre[:, q>block] += w_blk @ D11_blkᵀ (WMMA)
    const int firstQt = 2 * t + 2;
    const int ntiles  = (32 - firstQt) * 2;
    for (int ti = wid; ti < ntiles; ti += 8) {
      const int mt = ti & 1;
      const int qt = firstQt + (ti >> 1);
      const int rowb = mt * 16 + rh;
      v8f acc;
#pragma unroll
      for (int r = 0; r < 8; ++r) acc[r] = pre[(rowb + r) * PP + qt * 16 + ln];
      v16h a  = load_frag16x32(wsm,  PW,  mt * 16 + ln, kb, lane);
      v16h bb = load_frag16x32(D11h, NQD, qt * 16 + ln, kb, lane);
      acc = wmma_f16(a, bb, acc);
#pragma unroll
      for (int r = 0; r < 8; ++r) pre[(rowb + r) * PP + qt * 16 + ln] = acc[r];
    }
    // no barrier needed here: next iteration's stage+barrier orders pre-writes
  }

  // ---- outputs: dx (64 tiles) then y (32 tiles), distributed over 8 waves ----
  for (int tile = wid; tile < 96; tile += 8) {
    if (tile < 64) {
      const int mt = tile & 1, nt = tile >> 1;
      v8f acc = {};
      acc = gemm_acc(acc, xs,  PX, mt * 16 + ln, Ah,  NXD, nt * 16 + ln, NXD, lane);
      acc = gemm_acc(acc, wsm, PW, mt * 16 + ln, B1h, NQD, nt * 16 + ln, NQD, lane);
      acc = gemm_acc(acc, us,  PU, mt * 16 + ln, B2h, NUD, nt * 16 + ln, NUD, lane);
      const int rowb = b0 + mt * 16 + rh;
#pragma unroll
      for (int r = 0; r < 8; ++r)
        dx_out[(size_t)(rowb + r) * NXD + nt * 16 + ln] = acc[r];
    } else {
      const int tt = tile - 64;
      const int mt = tt & 1, nt = tt >> 1;
      v8f acc = {};
      acc = gemm_acc(acc, xs,  PX, mt * 16 + ln, C2h,  NXD, nt * 16 + ln, NXD, lane);
      acc = gemm_acc(acc, wsm, PW, mt * 16 + ln, D21h, NQD, nt * 16 + ln, NQD, lane);
      acc = gemm_acc(acc, us,  PU, mt * 16 + ln, D22h, NUD, nt * 16 + ln, NUD, lane);
      const int rowb = b0 + mt * 16 + rh;
#pragma unroll
      for (int r = 0; r < 8; ++r)
        y_out[(size_t)(rowb + r) * NYD + nt * 16 + ln] = acc[r];
    }
  }
}

extern "C" void kernel_launch(void* const* d_in, const int* in_sizes, int n_in,
                              void* d_out, int out_size, void* d_ws, size_t ws_size,
                              hipStream_t stream) {
  (void)in_sizes; (void)n_in; (void)out_size; (void)ws_size;
  const float* u   = (const float*)d_in[0];
  const float* x   = (const float*)d_in[1];
  const float* D12 = (const float*)d_in[2];
  const float* B2  = (const float*)d_in[3];
  const float* C2  = (const float*)d_in[4];
  const float* D21 = (const float*)d_in[5];
  const float* A   = (const float*)d_in[6];
  const float* D11 = (const float*)d_in[7];
  const float* C1  = (const float*)d_in[8];
  const float* B1  = (const float*)d_in[9];
  const float* D22 = (const float*)d_in[10];
  const float* Lam = (const float*)d_in[11];
  const float* bv  = (const float*)d_in[12];

  char* ws = (char*)d_ws;
  _Float16* C1h  = (_Float16*)(ws + 0);        // 512*512
  _Float16* D12h = (_Float16*)(ws + 524288);   // 512*256
  _Float16* D11h = (_Float16*)(ws + 786432);   // 512*512
  _Float16* Ah   = (_Float16*)(ws + 1310720);  // 512*512
  _Float16* B1h  = (_Float16*)(ws + 1835008);  // 512*512
  _Float16* B2h  = (_Float16*)(ws + 2359296);  // 512*256
  _Float16* C2h  = (_Float16*)(ws + 2621440);  // 256*512
  _Float16* D21h = (_Float16*)(ws + 2883584);  // 256*512
  _Float16* D22h = (_Float16*)(ws + 3145728);  // 256*256

  const int T = 256;
  auto cvt = [&](const float* s, _Float16* d, int n) {
    ren_cvt_f16<<<(n + T - 1) / T, T, 0, stream>>>(s, d, n);
  };
  cvt(C1,  C1h,  NQD * NXD);
  cvt(D12, D12h, NQD * NUD);
  cvt(D11, D11h, NQD * NQD);
  cvt(A,   Ah,   NXD * NXD);
  cvt(B1,  B1h,  NXD * NQD);
  cvt(B2,  B2h,  NXD * NUD);
  cvt(C2,  C2h,  NYD * NXD);
  cvt(D21, D21h, NYD * NQD);
  cvt(D22, D22h, NYD * NUD);

  float* dx_out = (float*)d_out;
  float* y_out  = dx_out + (size_t)NBATCH * NXD;

  const size_t smemBytes = 153728;   // pre + x/u/w tiles + D11 block + 1/Lambda
  ren_fused<<<NBATCH / BM, 256, smemBytes, stream>>>(
      u, x, C1h, D12h, D11h, Ah, B1h, B2h, C2h, D21h, D22h,
      D11, Lam, bv, dx_out, y_out);
}
